// MultiHeadAttention_39779987096034
// MI455X (gfx1250) — matrix operand order repaired and checked by reading, not compiled, not run on hardware
//
#include <hip/hip_runtime.h>

#define Bq 2
#define Sq 2048
#define Hq 1024
#define NHq 16
#define HDq 64
#define Mq (Bq*Sq)   // 4096

typedef __attribute__((ext_vector_type(16))) __bf16 v16bf;
typedef __attribute__((ext_vector_type(8)))  __bf16 v8bf;
typedef __attribute__((ext_vector_type(8)))  float  v8f;

__device__ __forceinline__ unsigned short f2bf_bits(float f) {
  union { float f; unsigned int u; } v; v.f = f;
  unsigned int r = (v.u + 0x7FFFu + ((v.u >> 16) & 1u)) >> 16;
  return (unsigned short)r;
}
__device__ __forceinline__ __bf16 f2bf(float f) {
  union { unsigned short s; __bf16 b; } o; o.s = f2bf_bits(f); return o.b;
}

// ---------------- fp32 -> bf16 convert ----------------
__global__ __launch_bounds__(256)
void mha_cvt_bf16(const float* __restrict__ in, unsigned short* __restrict__ out, int n) {
  int i = blockIdx.x * blockDim.x + threadIdx.x;
  int stride = gridDim.x * blockDim.x;
  for (; i < n; i += stride) out[i] = f2bf_bits(in[i]);
}

// ---------------- GEMM: Y[m,n] = sum_k A[m,k]*W[n,k] + bias[n] ----------------
// M=4096, N=K=1024. Block = 8 waves: 256 rows x 64 cols; wave = 32 rows x 64 cols.
// W tile (64x32 bf16, 4KB) staged in LDS (double-buffered), shared by all 8 waves.
// MODE 0: bf16 row-major; MODE 1: f32 row-major (d_out); MODE 2: bf16 V-transposed [B,NH,HD,S]
template<int MODE>
__global__ __launch_bounds__(256)
void mha_gemm_wmma(const __bf16* __restrict__ A, const __bf16* __restrict__ W,
                   const float* __restrict__ bias, void* __restrict__ Y)
{
  __shared__ __bf16 lds_w[2][64 * 32];   // 8 KB of the WGP's 320 KB

  const int lane = threadIdx.x & 31;
  const int g    = lane >> 4;
  const int l15  = lane & 15;
  const int wv   = threadIdx.x >> 5;     // wave 0..7 in block

  const int nstrip = blockIdx.x & 15;    // 16 strips of 64 columns
  const int mblk   = blockIdx.x >> 4;    // 16 row blocks of 256 rows
  const int mt     = mblk * 8 + wv;      // 32-row tile index (0..127)

  const __bf16* Arow0 = A + (size_t)(mt * 32 + l15) * Hq;       // rows 0-15 of tile
  const __bf16* Arow1 = A + (size_t)(mt * 32 + 16 + l15) * Hq;  // rows 16-31
  const __bf16* Wtile = W + (size_t)(nstrip * 64) * Hq;

  // LDS staging assignment: thread -> (row n_local = tid>>2, 16B chunk = tid&3)
  const int snl = threadIdx.x >> 2;
  const int skc = (threadIdx.x & 3) * 8;

  v8f acc[2][4];
#pragma unroll
  for (int s = 0; s < 2; ++s)
#pragma unroll
    for (int t = 0; t < 4; ++t) acc[s][t] = v8f{};

  for (int k0 = 0; k0 < Hq; k0 += 32) {
    const int cur = (k0 >> 5) & 1;
    // stage W[nstrip*64 .. +64, k0 .. k0+32) into LDS
    *(v8bf*)(&lds_w[cur][snl * 32 + skc]) =
        *(const v8bf*)(Wtile + (size_t)snl * Hq + k0 + skc);

    // prefetch next k-step's A rows while waiting
    if (k0 + 32 < Hq) {
      __builtin_prefetch((const void*)(Arow0 + k0 + 32), 0, 1);
      __builtin_prefetch((const void*)(Arow1 + k0 + 32), 0, 1);
    }
    __syncthreads();

    // A fragments (global): two 16-row sub-tiles
    v16bf a[2];
#pragma unroll
    for (int s = 0; s < 2; ++s) {
      const __bf16* Ar = s ? Arow1 : Arow0;
      v8bf lo = *(const v8bf*)(Ar + k0 + 16 * g);
      v8bf hi = *(const v8bf*)(Ar + k0 + 16 * g + 8);
#pragma unroll
      for (int j = 0; j < 8; ++j) { a[s][j] = lo[j]; a[s][j + 8] = hi[j]; }
    }

    // B fragments from LDS, reused across both A sub-tiles: 8 WMMAs / k-step
#pragma unroll
    for (int t = 0; t < 4; ++t) {
      v16bf bf = *(const v16bf*)(&lds_w[cur][(t * 16 + l15) * 32 + 16 * g]);
      acc[0][t] = __builtin_amdgcn_wmma_f32_16x16x32_bf16(false, a[0], false, bf,
                                                          (short)0, acc[0][t], false, false);
      acc[1][t] = __builtin_amdgcn_wmma_f32_16x16x32_bf16(false, a[1], false, bf,
                                                          (short)0, acc[1][t], false, false);
    }
    // double buffer: single barrier per step is sufficient
  }

#pragma unroll
  for (int t = 0; t < 4; ++t) {
    const int n = nstrip * 64 + t * 16 + l15;
    const float bv = bias[n];
#pragma unroll
    for (int s = 0; s < 2; ++s)
#pragma unroll
      for (int r = 0; r < 8; ++r) {
        const int mrow = mt * 32 + s * 16 + r + 8 * g;
        const float y = acc[s][t][r] + bv;
        if (MODE == 0) {
          ((unsigned short*)Y)[(size_t)mrow * Hq + n] = f2bf_bits(y);
        } else if (MODE == 1) {
          ((float*)Y)[(size_t)mrow * Hq + n] = y;
        } else {
          const int b = mrow >> 11, sidx = mrow & 2047;
          const int h = n >> 6,    d = n & 63;
          ((unsigned short*)Y)[((size_t)((b * NHq + h) * HDq + d)) * Sq + sidx] = f2bf_bits(y);
        }
      }
  }
}

// ---------------- flash attention (transposed-scores), 1 wave per (b,h,16q) ----------------
__global__ __launch_bounds__(256)
void mha_attn_wmma(const __bf16* __restrict__ Q,   // [B,S,H] bf16
                   const __bf16* __restrict__ Km,  // [B,S,H] bf16
                   const __bf16* __restrict__ Vt,  // [B,NH,HD,S] bf16
                   unsigned short* __restrict__ Ctx) // [B,S,H] bf16
{
  const int lane = threadIdx.x & 31;
  const int g    = lane >> 4;
  const int l15  = lane & 15;
  const int wid  = blockIdx.x * (blockDim.x >> 5) + (threadIdx.x >> 5);
  const int qt = wid & 127;
  const int h  = (wid >> 7) & 15;
  const int b  = wid >> 11;
  const int qbase = qt * 16;
  const float scale = 0.125f; // 1/sqrt(64)

  const __bf16* Qp = Q  + (size_t)(b * Sq + qbase) * Hq + h * HDq;
  const __bf16* Kp = Km + (size_t)b * Sq * Hq + h * HDq;
  const __bf16* Vp = Vt + (size_t)(b * NHq + h) * HDq * Sq;

  // Q as B-fragments (lane = query column); frag f covers head-dims [32f,32f+32)
  v16bf qb[2];
#pragma unroll
  for (int f = 0; f < 2; ++f)
    qb[f] = *(const v16bf*)(Qp + (size_t)l15 * Hq + f * 32 + g * 16);

  v8f acc[4] = {v8f{}, v8f{}, v8f{}, v8f{}};
  float mrun = -INFINITY, lrun = 0.f;

  for (int kv = 0; kv < Sq; kv += 32) {
    // prefetch next kv block's K rows and V columns
    if (kv + 32 < Sq) {
      __builtin_prefetch((const void*)(Kp + (size_t)(kv + 32 + l15) * Hq), 0, 1);
      __builtin_prefetch((const void*)(Kp + (size_t)(kv + 48 + l15) * Hq), 0, 1);
      __builtin_prefetch((const void*)(Vp + (size_t)(g * 32 + l15) * Sq + kv + 32), 0, 1);
    }

    // S^T tiles: rows = kv, cols = q. Two 16-kv tiles per 32-kv block.
    v8f s0 = {}, s1 = {};
#pragma unroll
    for (int t = 0; t < 2; ++t) {
      const __bf16* Krow = Kp + (size_t)(kv + t * 16 + l15) * Hq;
      v8f stile = (t == 0) ? s0 : s1;
#pragma unroll
      for (int f = 0; f < 2; ++f) {
        v8bf lo = *(const v8bf*)(Krow + f * 32 + 16 * g);
        v8bf hi = *(const v8bf*)(Krow + f * 32 + 16 * g + 8);
        v16bf a;
#pragma unroll
        for (int j = 0; j < 8; ++j) { a[j] = lo[j]; a[j + 8] = hi[j]; }
        stile = __builtin_amdgcn_wmma_f32_16x16x32_bf16(false, a, false, qb[f],
                                                        (short)0, stile, false, false);
      }
      if (t == 0) s0 = stile; else s1 = stile;
    }
#pragma unroll
    for (int r = 0; r < 8; ++r) { s0[r] *= scale; s1[r] *= scale; }

    // online softmax; lane owns query column q = l15 (lanes L and L^16 share q)
    float mloc = -INFINITY;
#pragma unroll
    for (int r = 0; r < 8; ++r) mloc = fmaxf(mloc, fmaxf(s0[r], s1[r]));
    mloc = fmaxf(mloc, __shfl_xor(mloc, 16, 32));
    const float mnew = fmaxf(mrun, mloc);
    const float alpha = __expf(mrun - mnew);

    float p0[8], p1[8], lloc = 0.f;
#pragma unroll
    for (int r = 0; r < 8; ++r) {
      p0[r] = __expf(s0[r] - mnew);
      p1[r] = __expf(s1[r] - mnew);
      lloc += p0[r] + p1[r];
    }
    lloc += __shfl_xor(lloc, 16, 32);
    lrun = lrun * alpha + lloc;
    mrun = mnew;

    // rescale O accumulator: acc row r belongs to query (r + 8g)
    float arow[8];
#pragma unroll
    for (int r = 0; r < 8; ++r) arow[r] = __shfl(alpha, 8 * g + r, 32);
#pragma unroll
    for (int f = 0; f < 4; ++f)
#pragma unroll
      for (int r = 0; r < 8; ++r) acc[f][r] *= arow[r];

    // P fragment is already in WMMA A-layout (lane=q row, K=kv)
    v16bf pa;
#pragma unroll
    for (int j = 0; j < 8; ++j) { pa[j] = f2bf(p0[j]); pa[j + 8] = f2bf(p1[j]); }

#pragma unroll
    for (int f = 0; f < 4; ++f) {
      v8bf vlo = *(const v8bf*)(Vp + (size_t)(f * 16 + l15) * Sq + kv + 8 * g);
      v8bf vhi = *(const v8bf*)(Vp + (size_t)(f * 16 + l15) * Sq + kv + 16 + 8 * g);
      v16bf vb;
#pragma unroll
      for (int j = 0; j < 8; ++j) { vb[j] = vlo[j]; vb[j + 8] = vhi[j]; }
      acc[f] = __builtin_amdgcn_wmma_f32_16x16x32_bf16(false, pa, false, vb,
                                                       (short)0, acc[f], false, false);
    }
  }

  const float linv = 1.0f / lrun;
  float lrow[8];
#pragma unroll
  for (int r = 0; r < 8; ++r) lrow[r] = __shfl(linv, 8 * g + r, 32);
#pragma unroll
  for (int f = 0; f < 4; ++f)
#pragma unroll
    for (int r = 0; r < 8; ++r) {
      const int qrow = qbase + r + 8 * g;
      Ctx[(size_t)(b * Sq + qrow) * Hq + h * HDq + f * 16 + l15] =
          f2bf_bits(acc[f][r] * lrow[r]);
    }
}

// ---------------- host launch ----------------
extern "C" void kernel_launch(void* const* d_in, const int* in_sizes, int n_in,
                              void* d_out, int out_size, void* d_ws, size_t ws_size,
                              hipStream_t stream) {
  (void)in_sizes; (void)n_in; (void)out_size; (void)ws_size;
  const float* query = (const float*)d_in[0];
  const float* key   = (const float*)d_in[1];
  const float* value = (const float*)d_in[2];
  const float* Wqf   = (const float*)d_in[3];
  const float* bqf   = (const float*)d_in[4];
  const float* Wkf   = (const float*)d_in[5];
  const float* bkf   = (const float*)d_in[6];
  const float* Wvf   = (const float*)d_in[7];
  const float* bvf   = (const float*)d_in[8];
  const float* Wof   = (const float*)d_in[9];
  const float* bof   = (const float*)d_in[10];

  char* ws = (char*)d_ws;
  const size_t MB = 1u << 20;
  unsigned short* xq = (unsigned short*)(ws + 0 * MB);   // 8 MB each activation
  unsigned short* xk = (unsigned short*)(ws + 8 * MB);
  unsigned short* xv = (unsigned short*)(ws + 16 * MB);
  unsigned short* wq = (unsigned short*)(ws + 24 * MB);  // 2 MB each weight
  unsigned short* wk = (unsigned short*)(ws + 26 * MB);
  unsigned short* wv = (unsigned short*)(ws + 28 * MB);
  unsigned short* wo = (unsigned short*)(ws + 30 * MB);
  unsigned short* Qp = (unsigned short*)(ws + 32 * MB);
  unsigned short* Kp = (unsigned short*)(ws + 40 * MB);
  unsigned short* Vt = (unsigned short*)(ws + 48 * MB);
  unsigned short* Ct = (unsigned short*)(ws + 56 * MB);

  const int nAct = Mq * Hq;     // 4M
  const int nWgt = Hq * Hq;     // 1M
  mha_cvt_bf16<<<2048, 256, 0, stream>>>(query, xq, nAct);
  mha_cvt_bf16<<<2048, 256, 0, stream>>>(key,   xk, nAct);
  mha_cvt_bf16<<<2048, 256, 0, stream>>>(value, xv, nAct);
  mha_cvt_bf16<<<512,  256, 0, stream>>>(Wqf, wq, nWgt);
  mha_cvt_bf16<<<512,  256, 0, stream>>>(Wkf, wk, nWgt);
  mha_cvt_bf16<<<512,  256, 0, stream>>>(Wvf, wv, nWgt);
  mha_cvt_bf16<<<512,  256, 0, stream>>>(Wof, wo, nWgt);

  // projections: 256 blocks x 8 waves, each wave 32x64 of the 4096x1024 output
  mha_gemm_wmma<0><<<256, 256, 0, stream>>>((const __bf16*)xq, (const __bf16*)wq, bqf, Qp);
  mha_gemm_wmma<0><<<256, 256, 0, stream>>>((const __bf16*)xk, (const __bf16*)wk, bkf, Kp);
  mha_gemm_wmma<2><<<256, 256, 0, stream>>>((const __bf16*)xv, (const __bf16*)wv, bvf, Vt);

  // attention: B*NH*(S/16) = 4096 waves -> 512 blocks
  mha_attn_wmma<<<512, 256, 0, stream>>>((const __bf16*)Qp, (const __bf16*)Kp,
                                         (const __bf16*)Vt, Ct);

  // output projection -> f32 d_out
  mha_gemm_wmma<1><<<256, 256, 0, stream>>>((const __bf16*)Ct, (const __bf16*)wo, bof,
                                            (float*)d_out);
}
